// ContrastiveGAE_87316685127955
// MI455X (gfx1250) — matrix operand order repaired and checked by reading, not compiled, not run on hardware
//
#include <hip/hip_runtime.h>
#include <hip/hip_fp16.h>

#define N_NODES 50000
#define N_EDGES 800000
#define ETOT (N_EDGES + N_NODES)
#define IN_CH 256
#define HID 64
#define HEADS 8
#define LAT 256
#define NUM_GRAPHS 64
#define NEG_SLOPE 0.2f

typedef _Float16 half_t;
typedef __attribute__((ext_vector_type(16))) _Float16 v16h;
typedef __attribute__((ext_vector_type(8)))  float    v8f;

union Frag16 { v16h v; uint4 q[2]; };
union FragC  { v8f  v; float f[8]; };

#define CEILDIV(a,b) (((a)+(b)-1)/(b))

// ---------------------------------------------------------------------------
// utility kernels
// ---------------------------------------------------------------------------
__global__ void fill_f32_k(float* __restrict__ p, float v, int n) {
  int i = blockIdx.x * blockDim.x + threadIdx.x;
  if (i < n) p[i] = v;
}

__global__ void cvt_f16_k(const float* __restrict__ in, half_t* __restrict__ out, int n) {
  int i = blockIdx.x * blockDim.x + threadIdx.x;
  if (i < n) out[i] = (half_t)in[i];
}

__device__ inline void atomicMaxF(float* addr, float v) {
  // sign-split monotone trick: signed max for >=0, unsigned min for <0
  if (v >= 0.0f) atomicMax((int*)addr, __float_as_int(v));
  else           atomicMin((unsigned int*)addr, __float_as_uint(v));
}

__device__ inline void atomicPkAddH2(half_t* addr, float f0, float f1) {
  // guarantee the native CDNA5 packed-f16 atomic (no CAS fallback):
  // GLOBAL_ATOMIC_PK_ADD_F16, no-return form (STOREcnt; s_endpgm waits idle)
  __half2 hv = __floats2half2_rn(f0, f1);
  unsigned int bits = *(unsigned int*)&hv;
  asm volatile("global_atomic_pk_add_f16 %0, %1, off"
               :: "v"((unsigned long long)(uintptr_t)addr), "v"(bits)
               : "memory");
}

__device__ inline void get_edge(const int* __restrict__ ei, int e, int& s, int& d) {
  if (e < N_EDGES) { s = ei[e]; d = ei[N_EDGES + e]; }
  else             { s = e - N_EDGES; d = s; }   // self loops appended
}

// ---------------------------------------------------------------------------
// Pack weight matrix B[K,Nc] (f32 row-major) into WMMA B-fragment layout:
// per (ntile, ktile): 32 lanes x 16 halfs; lane l holds N = nt*16+(l&15),
// K = kt*32 + (l>>4)*16 + [0..15].  One thread per packed dword.
// ---------------------------------------------------------------------------
__global__ void pack_b_k(const float* __restrict__ W, half_t* __restrict__ PB,
                         int K, int Nc)
{
  int i = blockIdx.x * blockDim.x + threadIdx.x;      // packed dword index
  if (i >= (K * Nc) / 2) return;
  int v    = i & 7;            // dword within lane chunk (0..7)
  int lane = (i >> 3) & 31;
  int t    = i >> 8;           // nt*KT + kt
  int KT   = K >> 5;
  int kt   = t % KT, nt = t / KT;
  int n  = nt * 16 + (lane & 15);
  int k0 = kt * 32 + (lane >> 4) * 16 + v * 2;
  PB[(size_t)i * 2 + 0] = (half_t)W[(size_t)(k0 + 0) * Nc + n];
  PB[(size_t)i * 2 + 1] = (half_t)W[(size_t)(k0 + 1) * Nc + n];
}

// ---------------------------------------------------------------------------
// LDS-free WMMA GEMM: C[M,Nc] = A[M,K] (f16 row-major) @ PB (pre-packed B).
// 256 threads = 8 waves; each wave owns one 16x16 output tile.
// A fragment = two coalesced-pairwise 16B chunks per lane straight from global.
// B fragment = two fully-coalesced global_load_b128 from the packed stream.
// Requires M%16==0, Nc%16==0, K%32==0.
// ---------------------------------------------------------------------------
__global__ __launch_bounds__(256)
void gemm_wmma_packed(const half_t* __restrict__ A, const half_t* __restrict__ PB,
                      const float* __restrict__ bias, float* __restrict__ Cf,
                      half_t* __restrict__ Ch, int M, int Nc, int K, int doRelu)
{
  const int lane = threadIdx.x & 31;
  const int wv   = threadIdx.x >> 5;
  const int NT   = Nc >> 4;
  const int KT   = K >> 5;
  const int tile = blockIdx.x * 8 + wv;
  if (tile >= (M >> 4) * NT) return;            // wave-uniform exit
  const int mt = tile / NT, nt = tile % NT;
  const int mrow   = lane & 15;
  const int halfid = lane >> 4;

  const half_t* ap = A + (size_t)(mt * 16 + mrow) * K + halfid * 16;
  const half_t* bp = PB + (size_t)nt * KT * 512 + lane * 16;

  v8f acc = {};
  #pragma unroll 2
  for (int kt = 0; kt < KT; ++kt) {
    Frag16 a, b;
    a.q[0] = *(const uint4*)(ap);
    a.q[1] = *(const uint4*)(ap + 8);
    b.q[0] = *(const uint4*)(bp);
    b.q[1] = *(const uint4*)(bp + 8);
    acc = __builtin_amdgcn_wmma_f32_16x16x32_f16(false, a.v, false, b.v,
                                                 (short)0, acc, false, false);
    ap += 32;
    bp += 512;
  }

  FragC c; c.v = acc;
  int ncol = nt * 16 + mrow;
  float bv = bias ? bias[ncol] : 0.0f;
  #pragma unroll
  for (int i = 0; i < 8; ++i) {
    int mr = mt * 16 + i + halfid * 8;   // VGPR i: lanes0-15 -> M=i, lanes16-31 -> M=i+8
    float val = c.f[i] + bv;
    if (doRelu) val = fmaxf(val, 0.0f);
    size_t o = (size_t)mr * Nc + ncol;
    if (Cf) Cf[o] = val;
    if (Ch) Ch[o] = (half_t)val;
  }
}

// ---------------------------------------------------------------------------
// attention scalar scores
// ---------------------------------------------------------------------------
__global__ void ascore8_k(const half_t* __restrict__ h, const float* __restrict__ ws,
                          const float* __restrict__ wd, float* __restrict__ asrc,
                          float* __restrict__ adst)
{
  int gtid = blockIdx.x * blockDim.x + threadIdx.x;
  int w = gtid >> 5;
  if (w >= N_NODES * HEADS) return;
  int lane = threadIdx.x & 31;
  int n = w >> 3, hd = w & 7;
  const half_t* p = h + (size_t)n * (HEADS * HID) + hd * HID;
  const float* s = ws + hd * HID;
  const float* d = wd + hd * HID;
  float s0 = (float)p[lane] * s[lane] + (float)p[lane + 32] * s[lane + 32];
  float d0 = (float)p[lane] * d[lane] + (float)p[lane + 32] * d[lane + 32];
  #pragma unroll
  for (int off = 16; off > 0; off >>= 1) {
    s0 += __shfl_xor(s0, off, 32);
    d0 += __shfl_xor(d0, off, 32);
  }
  if (lane == 0) { asrc[w] = s0; adst[w] = d0; }
}

__global__ void ascore1_k(const half_t* __restrict__ h, const float* __restrict__ ws,
                          const float* __restrict__ wd, float* __restrict__ asrc,
                          float* __restrict__ adst)
{
  int gtid = blockIdx.x * blockDim.x + threadIdx.x;
  int n = gtid >> 5;
  if (n >= N_NODES) return;
  int lane = threadIdx.x & 31;
  const half_t* p = h + (size_t)n * LAT;
  float s0 = 0.f, d0 = 0.f;
  #pragma unroll
  for (int j = 0; j < 8; ++j) {
    int c = lane + j * 32;
    float hv = (float)p[c];
    s0 += hv * ws[c];
    d0 += hv * wd[c];
  }
  #pragma unroll
  for (int off = 16; off > 0; off >>= 1) {
    s0 += __shfl_xor(s0, off, 32);
    d0 += __shfl_xor(d0, off, 32);
  }
  if (lane == 0) { asrc[n] = s0; adst[n] = d0; }
}

// ---------------------------------------------------------------------------
// edge pass A: leaky-relu logits + segment max over destination
// ---------------------------------------------------------------------------
__global__ void edge_max8_k(const int* __restrict__ ei, const float* __restrict__ asrc,
                            const float* __restrict__ adst, float* __restrict__ m)
{
  int i = blockIdx.x * blockDim.x + threadIdx.x;
  if (i >= ETOT * HEADS) return;
  int e = i >> 3, hd = i & 7;
  int s, d; get_edge(ei, e, s, d);
  float l = asrc[s * HEADS + hd] + adst[d * HEADS + hd];
  l = (l > 0.f) ? l : NEG_SLOPE * l;
  atomicMaxF(&m[d * HEADS + hd], l);
}

__global__ void edge_max1_k(const int* __restrict__ ei, const float* __restrict__ asrc,
                            const float* __restrict__ adst, float* __restrict__ m)
{
  int e = blockIdx.x * blockDim.x + threadIdx.x;
  if (e >= ETOT) return;
  int s, d; get_edge(ei, e, s, d);
  float l = asrc[s] + adst[d];
  l = (l > 0.f) ? l : NEG_SLOPE * l;
  atomicMaxF(&m[d], l);
}

// ---------------------------------------------------------------------------
// edge pass B: unnormalized softmax weights + message scatter via packed-f16
// atomics (GLOBAL_ATOMIC_PK_ADD_F16); wave per edge, lanes cover channel pairs
// ---------------------------------------------------------------------------
__global__ void edge_msg8_k(const int* __restrict__ ei, const float* __restrict__ asrc,
                            const float* __restrict__ adst, const float* __restrict__ m,
                            const half_t* __restrict__ h, half_t* __restrict__ outp,
                            float* __restrict__ den)
{
  int gtid = blockIdx.x * blockDim.x + threadIdx.x;
  int e = gtid >> 5;
  if (e >= ETOT) return;
  int lane = threadIdx.x & 31;
  int s, d; get_edge(ei, e, s, d);
  float my = 0.f;
  if (lane < HEADS) {
    float l = asrc[s * HEADS + lane] + adst[d * HEADS + lane];
    l = (l > 0.f) ? l : NEG_SLOPE * l;
    my = __expf(l - m[d * HEADS + lane]);
    atomicAdd(&den[d * HEADS + lane], my);
  }
  const __half2* hp = (const __half2*)(h + (size_t)s * (HEADS * HID));
  half_t* op = outp + (size_t)d * (HEADS * HID);
  #pragma unroll
  for (int it = 0; it < 8; ++it) {        // 256 half2 pairs / 32 lanes
    float eh = __shfl(my, it, 32);        // iter it covers exactly head it
    int c2 = lane + it * 32;              // half2 index
    float2 hv = __half22float2(hp[c2]);
    atomicPkAddH2(&op[c2 * 2], eh * hv.x, eh * hv.y);
  }
}

__global__ void edge_msg1_k(const int* __restrict__ ei, const float* __restrict__ asrc,
                            const float* __restrict__ adst, const float* __restrict__ m,
                            const half_t* __restrict__ h, half_t* __restrict__ outp,
                            float* __restrict__ den)
{
  int gtid = blockIdx.x * blockDim.x + threadIdx.x;
  int e = gtid >> 5;
  if (e >= ETOT) return;
  int lane = threadIdx.x & 31;
  int s, d; get_edge(ei, e, s, d);
  float my = 0.f;
  if (lane == 0) {
    float l = asrc[s] + adst[d];
    l = (l > 0.f) ? l : NEG_SLOPE * l;
    my = __expf(l - m[d]);
    atomicAdd(&den[d], my);
  }
  float eh = __shfl(my, 0, 32);
  const __half2* hp = (const __half2*)(h + (size_t)s * LAT);
  half_t* op = outp + (size_t)d * LAT;
  #pragma unroll
  for (int it = 0; it < 4; ++it) {        // 128 half2 pairs / 32 lanes
    int c2 = lane + it * 32;
    float2 hv = __half22float2(hp[c2]);
    atomicPkAddH2(&op[c2 * 2], eh * hv.x, eh * hv.y);
  }
}

// ---------------------------------------------------------------------------
// finalize: normalize accumulated messages, add bias (+relu layer 1)
// ---------------------------------------------------------------------------
__global__ void finalize1_k(const half_t* __restrict__ outp, const float* __restrict__ den,
                            const float* __restrict__ bias, half_t* __restrict__ g16)
{
  int i = blockIdx.x * blockDim.x + threadIdx.x;
  if (i >= N_NODES * HEADS * HID) return;
  int n = i >> 9, c = i & 511, hd = c >> 6;
  float v = (float)outp[i] / (den[n * HEADS + hd] + 1e-16f) + bias[c];
  v = fmaxf(v, 0.0f);
  g16[i] = (half_t)v;
}

__global__ void finalize2_k(const half_t* __restrict__ outp, const float* __restrict__ den,
                            const float* __restrict__ bias, float* __restrict__ z,
                            half_t* __restrict__ z16)
{
  int i = blockIdx.x * blockDim.x + threadIdx.x;
  if (i >= N_NODES * LAT) return;
  int n = i >> 8, c = i & 255;
  float v = (float)outp[i] / (den[n] + 1e-16f) + bias[c];
  z[i] = v;
  z16[i] = (half_t)v;
}

// ---------------------------------------------------------------------------
// pooling: batch is sorted -> per-block register accumulation, flush on
// graph-boundary.  thread = channel (256), block = 128 consecutive nodes.
// ---------------------------------------------------------------------------
__global__ void pool_count_k(const int* __restrict__ batch, float* __restrict__ counts)
{
  int n = blockIdx.x * blockDim.x + threadIdx.x;
  if (n < N_NODES) atomicAdd(&counts[batch[n]], 1.0f);
}

__global__ __launch_bounds__(256)
void pool_sum_k(const float* __restrict__ z, const int* __restrict__ batch,
                float* __restrict__ sums)
{
  int c = threadIdx.x;                 // channel 0..255
  int n0 = blockIdx.x * 128;
  int n1 = min(n0 + 128, N_NODES);
  float accum = 0.0f;
  int cur = batch[n0];
  for (int n = n0; n < n1; ++n) {
    int b = batch[n];                  // uniform across block
    if (b != cur) {
      atomicAdd(&sums[cur * LAT + c], accum);
      accum = 0.0f;
      cur = b;
    }
    accum += z[(size_t)n * LAT + c];
  }
  atomicAdd(&sums[cur * LAT + c], accum);
}

__global__ void graph_repr_k(const float* __restrict__ sums, const float* __restrict__ counts,
                             half_t* __restrict__ gr16)
{
  int i = blockIdx.x * blockDim.x + threadIdx.x;
  if (i >= NUM_GRAPHS * LAT) return;
  int g = i >> 8;
  gr16[i] = (half_t)(sums[i] / fmaxf(counts[g], 1.0f));
}

// ---------------------------------------------------------------------------
// launch
// ---------------------------------------------------------------------------
extern "C" void kernel_launch(void* const* d_in, const int* in_sizes, int n_in,
                              void* d_out, int out_size, void* d_ws, size_t ws_size,
                              hipStream_t stream)
{
  (void)in_sizes; (void)n_in; (void)out_size; (void)ws_size;
  const float* x     = (const float*)d_in[0];
  const int*   ei    = (const int*)  d_in[1];
  const int*   batch = (const int*)  d_in[2];
  const float* W1    = (const float*)d_in[3];
  const float* as1   = (const float*)d_in[4];
  const float* ad1   = (const float*)d_in[5];
  const float* b1    = (const float*)d_in[6];
  const float* W2    = (const float*)d_in[7];
  const float* as2   = (const float*)d_in[8];
  const float* ad2   = (const float*)d_in[9];
  const float* b2    = (const float*)d_in[10];
  const float* Pw1   = (const float*)d_in[11];
  const float* Pb1   = (const float*)d_in[12];
  const float* Pw2   = (const float*)d_in[13];
  const float* Pb2   = (const float*)d_in[14];
  float* out = (float*)d_out;

  char* base = (char*)d_ws;
  size_t off = 0;
  auto carve = [&](size_t bytes) -> void* {
    void* p = base + off;
    off += (bytes + 255) & ~(size_t)255;
    return p;
  };

  half_t* x16  = (half_t*)carve((size_t)N_NODES * IN_CH * 2);          // later z16
  half_t* PB1  = (half_t*)carve((size_t)IN_CH * HEADS * HID * 2);      // packed W1
  half_t* PB2  = (half_t*)carve((size_t)HEADS * HID * LAT * 2);        // packed W2
  half_t* PPw1 = (half_t*)carve((size_t)LAT * HID * 2);                // packed Pw1
  half_t* PPw2 = (half_t*)carve((size_t)HID * LAT * 2);                // packed Pw2
  half_t* h1h  = (half_t*)carve((size_t)N_NODES * HEADS * HID * 2);    // later out2 (f16)
  half_t* o1h  = (half_t*)carve((size_t)N_NODES * HEADS * HID * 2);    // f16 accumulators L1
  float*  zbuf = (float*) carve((size_t)N_NODES * LAT * 4);            // z (f32)
  half_t* g16  = (half_t*)carve((size_t)N_NODES * HEADS * HID * 2);    // later t1h
  half_t* h2h  = (half_t*)carve((size_t)N_NODES * LAT * 2);
  float* asrc1 = (float*)carve((size_t)N_NODES * HEADS * 4);
  float* adst1 = (float*)carve((size_t)N_NODES * HEADS * 4);
  float* m1    = (float*)carve((size_t)N_NODES * HEADS * 4);
  float* den1  = (float*)carve((size_t)N_NODES * HEADS * 4);
  float* asrc2 = (float*)carve((size_t)N_NODES * 4);
  float* adst2 = (float*)carve((size_t)N_NODES * 4);
  float* m2    = (float*)carve((size_t)N_NODES * 4);
  float* den2  = (float*)carve((size_t)N_NODES * 4);
  float* sums   = (float*)carve((size_t)NUM_GRAPHS * LAT * 4);
  float* counts = (float*)carve((size_t)NUM_GRAPHS * 4);
  half_t* gr16  = (half_t*)carve((size_t)NUM_GRAPHS * LAT * 2);
  half_t* tg16  = (half_t*)carve((size_t)NUM_GRAPHS * HID * 2);

  // aliases over dead regions
  half_t* o2h = h1h;        // layer-2 f16 accumulators; h1h dead after edge_msg8
  half_t* z16 = x16;        // x dead after GEMM1
  half_t* t1h = g16;        // g16 dead after GEMM2

  const int TB = 256;

  // ---- pack weights straight from f32 into WMMA-fragment layout ----
  pack_b_k<<<CEILDIV(IN_CH * HEADS * HID / 2, TB), TB, 0, stream>>>(W1, PB1, IN_CH, HEADS * HID);
  pack_b_k<<<CEILDIV(HEADS * HID * LAT / 2, TB), TB, 0, stream>>>(W2, PB2, HEADS * HID, LAT);
  pack_b_k<<<CEILDIV(LAT * HID / 2, TB), TB, 0, stream>>>(Pw1, PPw1, LAT, HID);
  pack_b_k<<<CEILDIV(HID * LAT / 2, TB), TB, 0, stream>>>(Pw2, PPw2, HID, LAT);
  cvt_f16_k<<<CEILDIV(N_NODES * IN_CH, TB), TB, 0, stream>>>(x, x16, N_NODES * IN_CH);

  // ---- layer 1: h1 = x @ W1 ----
  gemm_wmma_packed<<<CEILDIV((N_NODES / 16) * (HEADS * HID / 16), 8), 256, 0, stream>>>(
      x16, PB1, nullptr, nullptr, h1h, N_NODES, HEADS * HID, IN_CH, 0);

  ascore8_k<<<CEILDIV(N_NODES * HEADS * 32, TB), TB, 0, stream>>>(h1h, as1, ad1, asrc1, adst1);

  fill_f32_k<<<CEILDIV(N_NODES * HEADS, TB), TB, 0, stream>>>(m1, -3.0e38f, N_NODES * HEADS);
  fill_f32_k<<<CEILDIV(N_NODES * HEADS, TB), TB, 0, stream>>>(den1, 0.0f, N_NODES * HEADS);
  fill_f32_k<<<CEILDIV(N_NODES * HEADS * HID / 2, TB), TB, 0, stream>>>(
      (float*)o1h, 0.0f, N_NODES * HEADS * HID / 2);

  edge_max8_k<<<CEILDIV(ETOT * HEADS, TB), TB, 0, stream>>>(ei, asrc1, adst1, m1);
  edge_msg8_k<<<CEILDIV(ETOT * 32, TB), TB, 0, stream>>>(ei, asrc1, adst1, m1, h1h, o1h, den1);
  finalize1_k<<<CEILDIV(N_NODES * HEADS * HID, TB), TB, 0, stream>>>(o1h, den1, b1, g16);

  // ---- layer 2: h2 = g @ W2 ----
  gemm_wmma_packed<<<CEILDIV((N_NODES / 16) * (LAT / 16), 8), 256, 0, stream>>>(
      g16, PB2, nullptr, nullptr, h2h, N_NODES, LAT, HEADS * HID, 0);

  ascore1_k<<<CEILDIV(N_NODES * 32, TB), TB, 0, stream>>>(h2h, as2, ad2, asrc2, adst2);

  fill_f32_k<<<CEILDIV(N_NODES, TB), TB, 0, stream>>>(m2, -3.0e38f, N_NODES);
  fill_f32_k<<<CEILDIV(N_NODES, TB), TB, 0, stream>>>(den2, 0.0f, N_NODES);
  fill_f32_k<<<CEILDIV(N_NODES * LAT / 2, TB), TB, 0, stream>>>(
      (float*)o2h, 0.0f, N_NODES * LAT / 2);

  edge_max1_k<<<CEILDIV(ETOT, TB), TB, 0, stream>>>(ei, asrc2, adst2, m2);
  edge_msg1_k<<<CEILDIV(ETOT * 32, TB), TB, 0, stream>>>(ei, asrc2, adst2, m2, h2h, o2h, den2);
  finalize2_k<<<CEILDIV(N_NODES * LAT, TB), TB, 0, stream>>>(o2h, den2, b2, zbuf, z16);

  // ---- global mean pool (sorted batch -> register accumulation) ----
  fill_f32_k<<<CEILDIV(NUM_GRAPHS * LAT, TB), TB, 0, stream>>>(sums, 0.0f, NUM_GRAPHS * LAT);
  fill_f32_k<<<1, TB, 0, stream>>>(counts, 0.0f, NUM_GRAPHS);
  pool_count_k<<<CEILDIV(N_NODES, TB), TB, 0, stream>>>(batch, counts);
  pool_sum_k<<<CEILDIV(N_NODES, 128), 256, 0, stream>>>(zbuf, batch, sums);
  graph_repr_k<<<CEILDIV(NUM_GRAPHS * LAT, TB), TB, 0, stream>>>(sums, counts, gr16);

  // ---- projection MLP: relu(t@Pw1+Pb1)@Pw2+Pb2 ----
  gemm_wmma_packed<<<CEILDIV((N_NODES / 16) * (HID / 16), 8), 256, 0, stream>>>(
      z16, PPw1, Pb1, nullptr, t1h, N_NODES, HID, LAT, 1);
  gemm_wmma_packed<<<CEILDIV((N_NODES / 16) * (LAT / 16), 8), 256, 0, stream>>>(
      t1h, PPw2, Pb2, out, nullptr, N_NODES, LAT, HID, 0);

  gemm_wmma_packed<<<CEILDIV((NUM_GRAPHS / 16) * (HID / 16), 8), 256, 0, stream>>>(
      gr16, PPw1, Pb1, nullptr, tg16, NUM_GRAPHS, HID, LAT, 1);
  gemm_wmma_packed<<<CEILDIV((NUM_GRAPHS / 16) * (LAT / 16), 8), 256, 0, stream>>>(
      tg16, PPw2, Pb2, out + (size_t)N_NODES * LAT, nullptr, NUM_GRAPHS, LAT, HID, 0);
}
